// EALSTM_66700842107128
// MI455X (gfx1250) — matrix operand order repaired and checked by reading, compiled, not run on hardware
//
#include <hip/hip_runtime.h>

#define B_    1024
#define T_    365
#define D_    5
#define S_    27
#define H_    256
#define DH    261      // D_ + H_
#define KAUG  288      // 256 (h) + 5 (x) + 1 (bias) + 26 pad  -> 9 k-fragments of 32
#define NKF   9
#define NKRES 7        // kf 0..6 register-resident; kf 7..8 streamed from LDS
#define BT    16       // batch rows per workgroup
#define LDSROW 296     // padded LDS row stride (bf16 elems), 16B-aligned stride
#define NTHREADS 128   // 4 waves, one per SIMD: wave w owns N-tiles 4w..4w+3
#define NTPW  4        // N-tiles per wave
#define NCOMBO (3 * 16)            // gate x n_tile combos per workgroup
#define NKLDS (NKF - NKRES)        // 2 k-fragments streamed from LDS
#define BLDSELEMS (NKLDS * NCOMBO * 512)   // 96 KB of bf16 in LDS

typedef __attribute__((ext_vector_type(16))) __bf16 v16bf;
typedef __attribute__((ext_vector_type(8)))  float  v8f;

union Frag {
    v16bf v;
    uint4 q[2];
};

// Fast activations: one v_exp + one v_rcp each (avoid IEEE div sequences).
__device__ __forceinline__ float sigmoid_f(float x) {
    return __builtin_amdgcn_rcpf(1.0f + __expf(-x));
}
__device__ __forceinline__ float tanh_f(float x) {
    return 1.0f - 2.0f * __builtin_amdgcn_rcpf(__expf(2.0f * x) + 1.0f);
}

// ---------------------------------------------------------------------------
// Prep: pack augmented weight rows [W_h | W_x | bias | 0] (K=288) into
// per-lane-contiguous bf16 WMMA B-fragments:
//   bpack[gate][n_tile][k_frag][lane][16]
// B-matrix (32x16) fragment layout: lane L holds column N = (L&15); its 16
// bf16 values are K = kf*32 + (L&16) + j, j=0..15.
// ---------------------------------------------------------------------------
__global__ void ealstm_pack(const float* __restrict__ Wf, const float* __restrict__ bf,
                            const float* __restrict__ Wg, const float* __restrict__ bg,
                            const float* __restrict__ Wo, const float* __restrict__ bo,
                            __bf16* __restrict__ bpack)
{
    int idx  = blockIdx.x * blockDim.x + threadIdx.x;   // 3*16*9*512 total
    int j    = idx & 15;
    int lane = (idx >> 4) & 31;
    int frag = idx >> 9;                 // 0..431
    int kf   = frag % NKF;
    int nt   = (frag / NKF) & 15;
    int g    = frag / (NKF * 16);

    int k = kf * 32 + (lane & 16) + j;   // augmented K index
    int n = nt * 16 + (lane & 15);       // output-H index

    const float* W  = (g == 0) ? Wf : (g == 1) ? Wg : Wo;
    const float* bb = (g == 0) ? bf : (g == 1) ? bg : bo;

    float v = 0.0f;
    if (k < H_)            v = W[n * DH + D_ + k];      // recurrent part  W[:, D:]
    else if (k < H_ + D_)  v = W[n * DH + (k - H_)];    // input part      W[:, :D]
    else if (k == H_ + D_) v = bb[n];                   // bias row (pairs with 1.0)
    bpack[idx] = (__bf16)v;
}

// ---------------------------------------------------------------------------
// Main: one workgroup per 16 batch rows; 4 waves (1 per SIMD32, full
// 1024-VGPR budget each). Wave w owns N-tiles 4w..4w+3 for ALL three gates
// -> elementwise update is lane-local. B fragments kf=0..6 (84 frags =
// 672 VGPRs) stay register-resident; kf=7..8 live in LDS and are re-read
// each step (leaves ~150 VGPRs of scheduler headroom -> no spills).
// A fragments double-buffered; x(t+1) prefetched during step t.
// ---------------------------------------------------------------------------
__global__ __launch_bounds__(NTHREADS) void ealstm_main(
    const float* __restrict__ x_dyn,  const float* __restrict__ x_stat,
    const float* __restrict__ W_i,    const float* __restrict__ b_i,
    const float* __restrict__ W_head, const float* __restrict__ b_head,
    const __bf16* __restrict__ bpack, float* __restrict__ out)
{
    __shared__ alignas(32) __bf16 hbuf[BT][LDSROW];     // 9.25 KB
    __shared__ alignas(32) __bf16 blds[BLDSELEMS];      // 96 KB: [kf-7][combo][512]

    const int tid  = threadIdx.x;
    const int lane = tid & 31;
    const int wave = tid >> 5;       // 0..3
    const int b0   = blockIdx.x * BT;
    const int lhi  = lane >> 4;      // 0 or 1
    const int l15  = lane & 15;

    // init LDS h-buffer: h=0, x cols=0, col (H_+D_)=1.0 (bias mult), pad 0
    __bf16* hb = &hbuf[0][0];
    for (int i = tid; i < BT * LDSROW; i += NTHREADS) {
        int col = i % LDSROW;
        hb[i] = (col == (H_ + D_)) ? (__bf16)1.0f : (__bf16)0.0f;
    }

    // copy kf=7..8 fragments for all 48 combos into LDS (uint4 granules)
    {
        uint4* dst = (uint4*)blds;
        for (int i = tid; i < BLDSELEMS / 8; i += NTHREADS) {
            int q  = i & 63;                 // uint4 index within fragment
            int c  = (i >> 6) % NCOMBO;      // combo = g*16 + nt
            int kk = i / (64 * NCOMBO);      // 0 or 1  (kf = NKRES + kk)
            const uint4* src = (const uint4*)
                (bpack + (((size_t)(c * NKF + NKRES + kk)) << 9));
            dst[i] = src[q];
        }
    }

    // Hoist this wave's 84 resident B fragments (kf=0..6, 12 combos).
    Frag bw[3][NTPW][NKRES];
    #pragma unroll
    for (int g = 0; g < 3; ++g) {
        #pragma unroll
        for (int j = 0; j < NTPW; ++j) {
            const int nt = wave * NTPW + j;
            #pragma unroll
            for (int kf = 0; kf < NKRES; ++kf) {
                const uint4* bp = (const uint4*)
                    (bpack + (((size_t)((g * 16 + nt) * NKF + kf)) << 9)
                           + lane * 16);
                bw[g][j][kf].q[0] = bp[0];
                bw[g][j][kf].q[1] = bp[1];
            }
        }
    }

    // i_gate (once) + c-state in C/D layout:
    // element (j, r) <-> row b = b0 + r + 8*lhi, col n = (wave*4+j)*16 + l15
    float cst[NTPW][8];
    float ig [NTPW][8];
    #pragma unroll
    for (int j = 0; j < NTPW; ++j) {
        const int n = (wave * NTPW + j) * 16 + l15;
        for (int r = 0; r < 8; ++r) {
            int b = b0 + r + lhi * 8;
            float s = b_i[n];
            for (int si = 0; si < S_; ++si)
                s += x_stat[b * S_ + si] * W_i[n * S_ + si];
            ig [j][r] = sigmoid_f(s);
            cst[j][r] = 0.0f;
        }
    }

    // prefetch x(t=0) into a register (threads 0..79 own one (row, d) slot)
    const int xr = tid / D_;         // row 0..15 (valid when tid < 80)
    const int xd = tid % D_;
    float xreg = 0.0f;
    if (tid < BT * D_)
        xreg = x_dyn[((size_t)(b0 + xr) * T_ + 0) * D_ + xd];

    __syncthreads();

    const v8f vzero = {};

    for (int t = 0; t < T_; ++t) {
        // keep per-step LDS reads (blds) from being hoisted into registers
        asm volatile("" ::: "memory");

        // store prefetched x_t into augmented columns 256..260
        if (tid < BT * D_)
            hbuf[xr][H_ + xd] = (__bf16)xreg;
        __syncthreads();   // h_{t-1} and x_t visible to all waves

        // prefetch x(t+1): global latency hidden behind this step's compute
        if (tid < BT * D_) {
            int tn = (t + 1 < T_) ? (t + 1) : t;
            xreg = x_dyn[((size_t)(b0 + xr) * T_ + tn) * D_ + xd];
        }

        v8f acc[3][NTPW];
        #pragma unroll
        for (int g = 0; g < 3; ++g)
            #pragma unroll
            for (int j = 0; j < NTPW; ++j)
                acc[g][j] = vzero;

        // A fragments double-buffered: kf+1 loads issue under kf's WMMAs.
        Frag a[2];
        a[0].q[0] = *(const uint4*)&hbuf[l15][lhi * 16];
        a[0].q[1] = *(const uint4*)&hbuf[l15][lhi * 16 + 8];

        #pragma unroll
        for (int kf = 0; kf < NKF; ++kf) {
            if (kf + 1 < NKF) {
                const int koff = (kf + 1) * 32 + lhi * 16;
                a[(kf + 1) & 1].q[0] = *(const uint4*)&hbuf[l15][koff];
                a[(kf + 1) & 1].q[1] = *(const uint4*)&hbuf[l15][koff + 8];
            }
            if (kf < NKRES) {
                // B register-resident
                #pragma unroll
                for (int j = 0; j < NTPW; ++j)
                    #pragma unroll
                    for (int g = 0; g < 3; ++g)
                        acc[g][j] = __builtin_amdgcn_wmma_f32_16x16x32_bf16(
                            false, a[kf & 1].v, false, bw[g][j][kf].v,
                            (short)0, acc[g][j], false, false);
            } else {
                // kf=7..8: B streamed from LDS
                #pragma unroll
                for (int j = 0; j < NTPW; ++j) {
                    const int nt = wave * NTPW + j;
                    #pragma unroll
                    for (int g = 0; g < 3; ++g) {
                        const uint4* bp = (const uint4*)
                            &blds[((kf - NKRES) * NCOMBO + g * 16 + nt) * 512
                                  + lane * 16];
                        Frag bs;
                        bs.q[0] = bp[0];
                        bs.q[1] = bp[1];
                        acc[g][j] = __builtin_amdgcn_wmma_f32_16x16x32_bf16(
                            false, a[kf & 1].v, false, bs.v,
                            (short)0, acc[g][j], false, false);
                    }
                }
            }
        }
        __syncthreads();   // all A reads complete before h is overwritten

        // lane-local gate math + state update; write h_t back to LDS (bf16)
        #pragma unroll
        for (int j = 0; j < NTPW; ++j) {
            #pragma unroll
            for (int r = 0; r < 8; ++r) {
                float fg = sigmoid_f(acc[0][j][r]);
                float gg = tanh_f  (acc[1][j][r]);
                float og = sigmoid_f(acc[2][j][r]);
                float c  = fg * cst[j][r] + ig[j][r] * gg;
                cst[j][r] = c;
                float h  = og * tanh_f(c);
                hbuf[r + lhi * 8][(wave * NTPW + j) * 16 + l15] = (__bf16)h;
            }
        }
    }
    __syncthreads();

    // head: out[b] = h_T . W_head + b_head
    if (tid < BT) {
        float s = 0.0f;
        for (int n = 0; n < H_; ++n)
            s += (float)hbuf[tid][n] * W_head[n];
        out[b0 + tid] = s + b_head[0];
    }
}

// ---------------------------------------------------------------------------
extern "C" void kernel_launch(void* const* d_in, const int* in_sizes, int n_in,
                              void* d_out, int out_size, void* d_ws, size_t ws_size,
                              hipStream_t stream)
{
    const float* x_dyn  = (const float*)d_in[0];
    const float* x_stat = (const float*)d_in[1];
    const float* W_i    = (const float*)d_in[2];
    const float* b_i    = (const float*)d_in[3];
    const float* W_f    = (const float*)d_in[4];
    const float* b_f    = (const float*)d_in[5];
    const float* W_g    = (const float*)d_in[6];
    const float* b_g    = (const float*)d_in[7];
    const float* W_o    = (const float*)d_in[8];
    const float* b_o    = (const float*)d_in[9];
    const float* W_head = (const float*)d_in[10];
    const float* b_head = (const float*)d_in[11];

    __bf16* bpack = (__bf16*)d_ws;              // 3*16*9*512 bf16 = 432 KB
    float*  out   = (float*)d_out;

    const int packElems = 3 * 16 * NKF * 512;   // 221184
    ealstm_pack<<<packElems / 256, 256, 0, stream>>>(W_f, b_f, W_g, b_g,
                                                     W_o, b_o, bpack);
    ealstm_main<<<B_ / BT, NTHREADS, 0, stream>>>(x_dyn, x_stat, W_i, b_i,
                                                  W_head, b_head, bpack, out);
}
